// GraphNet_1726576854583
// MI455X (gfx1250) — matrix operand order repaired and checked by reading, not compiled, not run on hardware
//
#include <hip/hip_runtime.h>
#include <hip/hip_bf16.h>

// ---------------------------------------------------------------------------
// Types for CDNA5 WMMA (wave32): A/B = 16 x bf16, C/D = 8 x f32 per lane.
// ---------------------------------------------------------------------------
typedef __attribute__((ext_vector_type(16))) __bf16   v16bf;
typedef __attribute__((ext_vector_type(8)))  float    v8f;
typedef __attribute__((ext_vector_type(4)))  uint32_t u32x4;
typedef __attribute__((ext_vector_type(8)))  uint32_t u32x8;

#define N1 2048
#define N2 2048
#define DD 512
#define QQ 2048
#define GG 2048
#define AA 4096

// Generic-pointer -> LDS byte offset (AS3 pointer value IS the LDS offset).
__device__ __forceinline__ uint32_t lds_off(const float* p)
{
    return (uint32_t)(uintptr_t)(__attribute__((address_space(3))) const float*)p;
}

// ---------------------------------------------------------------------------
// Tensor Data Mover: one 2D-tile DMA global->LDS (TENSORcnt-tracked).
// D# packing per CDNA5 ISA (group0) / §8.4 (group1):
//   group0: [1:0]=count=1 | [63:32]=lds_addr | [120:64]=global_addr | [127:126]=type=2
//   group1: [15:0]=wg_mask=0 | [17:16]=data_size=2 (4B) | [79:48]=tensor_dim0
//           [111:80]=tensor_dim1 | [127:112]=tile_dim0 | [143:128]=tile_dim1
//           [159:144]=tile_dim2=0 | [207:160]=tensor_dim0_stride
// Tile rows land packed back-to-back in LDS (row length = tile_w elems).
// tensor_dim* set equal to tile dims (tiles never go OOB here).
// ---------------------------------------------------------------------------
__device__ __forceinline__ void tdm_load_2d(uint32_t lds_addr, const float* gaddr,
                                            uint32_t tile_w, uint32_t tile_h,
                                            uint32_t stride_elems)
{
    const uint64_t ga = (uint64_t)(uintptr_t)gaddr;
    u32x4 g0;
    g0[0] = 1u;                                              // count=1 (valid user D#)
    g0[1] = lds_addr;                                        // LDS byte address
    g0[2] = (uint32_t)ga;                                    // global addr [31:0]
    g0[3] = (uint32_t)((ga >> 32) & 0x01FFFFFFu) | (2u << 30); // addr [56:32] | type=2
    u32x8 g1;
    g1[0] = 0x00020000u;                                     // data_size=2 (4 bytes)
    g1[1] = (tile_w & 0xFFFFu) << 16;                        // tensor_dim0 lo16
    g1[2] = (tile_w >> 16) | ((tile_h & 0xFFFFu) << 16);     // dim0 hi16 | dim1 lo16
    g1[3] = (tile_h >> 16) | ((tile_w & 0xFFFFu) << 16);     // dim1 hi16 | tile_dim0
    g1[4] = tile_h & 0xFFFFu;                                // tile_dim1 | tile_dim2=0
    g1[5] = stride_elems;                                    // tensor_dim0_stride lo32
    g1[6] = 0u;                                              // stride hi16 | dim1_stride lo
    g1[7] = 0u;
    asm volatile("tensor_load_to_lds %0, %1" :: "s"(g0), "s"(g1) : "memory");
}

__device__ __forceinline__ void wait_tensor0()
{
    asm volatile("s_wait_tensorcnt 0x0" ::: "memory");
}

// ---------------------------------------------------------------------------
// bf16x3 split-precision WMMA GEMM with TDM-staged, double-buffered LDS tiles.
//   D = op(A)*op(B) [+C] [+bias] [relu]
// Block: 256 threads = 8 waves (4x2). Wave: 2x2 tiles of 16x16 -> 32x32.
// Block tile: 128(M) x 64(N) x 32(K). All problem dims divide evenly.
//
// LDS layouts == TDM row-packed tile layouts:
//   TA=1: As[k][m] (32 rows x 128)   TA=0: As[m][k] (128 rows x 32)
//   TB=1: Bs[n][k] (64 rows x 32)    TB=0: Bs[k][n] (32 rows x 64)
//
// bf16 A fragment (16x32): lane L: m=L%16, half=L/16;
//   e in [0,8): k=half*8+e ; e in [8,16): k=16+half*8+(e-8)
// bf16 B fragment (32x16): lane L: n=L%16, half=L/16; same k order as A
// f32 C/D (16x16): lane L: n=L%16; VGPR r holds m=r+(L/16)*8
// ---------------------------------------------------------------------------
template<int TA, int TB>
__device__ __forceinline__ void stage_tiles(const float* __restrict__ A, int lda,
                                            const float* __restrict__ B, int ldb,
                                            int mblk, int nblk, int kb,
                                            float* as, float* bs)
{
    if (TA) tdm_load_2d(lds_off(as), A + (size_t)kb * lda + mblk, 128, 32, (uint32_t)lda);
    else    tdm_load_2d(lds_off(as), A + (size_t)mblk * lda + kb,  32, 128, (uint32_t)lda);
    if (TB) tdm_load_2d(lds_off(bs), B + (size_t)nblk * ldb + kb,  32,  64, (uint32_t)ldb);
    else    tdm_load_2d(lds_off(bs), B + (size_t)kb * ldb + nblk,  64,  32, (uint32_t)ldb);
}

template<int TA, int TB, int RELU, int ADDC, int BIAS>
__global__ __launch_bounds__(256)
void wmma_gemm(const float* __restrict__ A, int lda,
               const float* __restrict__ B, int ldb,
               const float* __restrict__ C, int ldc,
               const float* __restrict__ bias,
               float* __restrict__ D, int ldd, int K)
{
    __shared__ float As[2][32 * 128];   // 16 KB each
    __shared__ float Bs[2][32 * 64];    //  8 KB each   (48 KB total)

    const int tid  = threadIdx.x;
    const int lane = tid & 31;
    const int wave = tid >> 5;
    const int l16  = lane & 15;
    const int hf   = lane >> 4;
    const int wm   = wave >> 1;            // 0..3
    const int wn   = wave & 1;             // 0..1
    const int mblk = blockIdx.y * 128;
    const int nblk = blockIdx.x * 64;

    v8f acc[2][2];
    #pragma unroll
    for (int mt = 0; mt < 2; ++mt)
        #pragma unroll
        for (int nt = 0; nt < 2; ++nt)
            acc[mt][nt] = (v8f)(0.0f);

    if (wave == 0)                      // one wave drives the TDM
        stage_tiles<TA, TB>(A, lda, B, ldb, mblk, nblk, 0, As[0], Bs[0]);

    int cur = 0;
    for (int kb = 0; kb < K; kb += 32) {
        if (wave == 0) wait_tensor0();  // issuing wave sees TDM completion
        __syncthreads();                // ...and publishes it to all waves

        if (kb + 32 < K && wave == 0)
            stage_tiles<TA, TB>(A, lda, B, ldb, mblk, nblk, kb + 32,
                                As[cur ^ 1], Bs[cur ^ 1]);

        const float* as = As[cur];
        const float* bs = Bs[cur];

        v16bf ahi[2], alo[2], bhi[2], blo[2];
        #pragma unroll
        for (int mt = 0; mt < 2; ++mt) {
            const int m = wm * 32 + mt * 16 + l16;
            #pragma unroll
            for (int e = 0; e < 16; ++e) {
                const int k = (e < 8) ? hf * 8 + e : 16 + hf * 8 + (e - 8);
                const float x = TA ? as[k * 128 + m] : as[m * 32 + k];
                const __bf16 h = (__bf16)x;
                ahi[mt][e] = h;
                alo[mt][e] = (__bf16)(x - (float)h);
            }
        }
        #pragma unroll
        for (int nt = 0; nt < 2; ++nt) {
            const int n = wn * 32 + nt * 16 + l16;
            #pragma unroll
            for (int e = 0; e < 16; ++e) {
                const int k = (e < 8) ? hf * 8 + e : 16 + hf * 8 + (e - 8);
                const float x = TB ? bs[n * 32 + k] : bs[k * 64 + n];
                const __bf16 h = (__bf16)x;
                bhi[nt][e] = h;
                blo[nt][e] = (__bf16)(x - (float)h);
            }
        }
        #pragma unroll
        for (int mt = 0; mt < 2; ++mt)
            #pragma unroll
            for (int nt = 0; nt < 2; ++nt) {
                acc[mt][nt] = __builtin_amdgcn_wmma_f32_16x16x32_bf16(
                    false, ahi[mt], false, bhi[nt], (short)0, acc[mt][nt], false, false);
                acc[mt][nt] = __builtin_amdgcn_wmma_f32_16x16x32_bf16(
                    false, ahi[mt], false, blo[nt], (short)0, acc[mt][nt], false, false);
                acc[mt][nt] = __builtin_amdgcn_wmma_f32_16x16x32_bf16(
                    false, alo[mt], false, bhi[nt], (short)0, acc[mt][nt], false, false);
            }

        __syncthreads();        // all waves done reading buffer `cur`
        cur ^= 1;
    }

    #pragma unroll
    for (int mt = 0; mt < 2; ++mt)
        #pragma unroll
        for (int nt = 0; nt < 2; ++nt)
            #pragma unroll
            for (int r = 0; r < 8; ++r) {
                const int m = mblk + wm * 32 + mt * 16 + r + hf * 8;
                const int n = nblk + wn * 32 + nt * 16 + l16;
                float v = acc[mt][nt][r];
                if (ADDC) v += C[(size_t)m * ldc + n];
                if (BIAS) v += bias[n];
                if (RELU) v = fmaxf(v, 0.0f);
                D[(size_t)m * ldd + n] = v;
            }
}

// ---------------------------------------------------------------------------
// l[i] = ||U[:,i]|| / ||Mrow[i,:]||   (U is D x N, Mrow is N x D)
// ---------------------------------------------------------------------------
__global__ void compute_l(const float* __restrict__ U, const float* __restrict__ Mrow,
                          float* __restrict__ l, int N, int D)
{
    const int i = blockIdx.x * blockDim.x + threadIdx.x;
    if (i >= N) return;
    float s1 = 0.f, s2 = 0.f;
    for (int d = 0; d < D; ++d) {
        const float u = U[(size_t)d * N + i]; s1 += u * u;
        const float m = Mrow[(size_t)i * D + d]; s2 += m * m;
    }
    l[i] = sqrtf(s1) / sqrtf(s2);
}

// M[i,d] = U[d,i] + l[i]*M[i,d]   (in place over M)
__global__ void build_x(const float* __restrict__ U, const float* __restrict__ l,
                        float* __restrict__ M, int N, int D)
{
    const size_t idx = (size_t)blockIdx.x * blockDim.x + threadIdx.x;
    if (idx >= (size_t)N * D) return;
    const int i = (int)(idx / D);
    const int d = (int)(idx % D);
    M[idx] = U[(size_t)d * N + i] + l[i] * M[idx];
}

// row-wise L2 normalize with eps clamp
__global__ void row_norm(float* __restrict__ E, int R, int D)
{
    const int r = blockIdx.x * blockDim.x + threadIdx.x;
    if (r >= R) return;
    float s = 0.f;
    for (int d = 0; d < D; ++d) { const float v = E[(size_t)r * D + d]; s += v * v; }
    const float inv = 1.0f / fmaxf(sqrtf(s), 1e-12f);
    for (int d = 0; d < D; ++d) E[(size_t)r * D + d] *= inv;
}

// ---------------------------------------------------------------------------
// Per row of S (A=4096): stable top-2 argmax (== argsort of d ascending since
// d=(2-2s)^2/rowmax is monotone-decreasing in s) + row min for the column-max
// normalizer (S symmetric => colmax(dd) == rowmax(dd) = (2-2*min s)^2).
// ---------------------------------------------------------------------------
__global__ __launch_bounds__(256)
void row_top2(const float* __restrict__ S, int A,
              int* __restrict__ n0, int* __restrict__ n1, float* __restrict__ rowmax)
{
    __shared__ float s1[256]; __shared__ int j1[256];
    __shared__ float s2[256]; __shared__ int j2[256];
    __shared__ float mn[256];
    const int tid = threadIdx.x;
    const int row = blockIdx.x;
    const float* Sr = S + (size_t)row * A;

    float b1 = -3.402823466e38f, b2 = -3.402823466e38f;
    int   i1 = 0x7fffffff,       i2 = 0x7fffffff;
    float smin = 3.402823466e38f;
    for (int j = tid; j < A; j += 256) {
        const float s = Sr[j];
        smin = fminf(smin, s);
        const bool bt1 = (s > b1) || (s == b1 && j < i1);
        const bool bt2 = (s > b2) || (s == b2 && j < i2);
        if (bt1)      { b2 = b1; i2 = i1; b1 = s; i1 = j; }
        else if (bt2) { b2 = s; i2 = j; }
    }
    s1[tid] = b1; j1[tid] = i1; s2[tid] = b2; j2[tid] = i2; mn[tid] = smin;
    __syncthreads();
    for (int off = 128; off > 0; off >>= 1) {
        if (tid < off) {
            const float q1 = s1[tid + off]; const int qi1 = j1[tid + off];
            const float q2 = s2[tid + off]; const int qi2 = j2[tid + off];
            const float p1 = s1[tid];       const int pi1 = j1[tid];
            const float p2 = s2[tid];       const int pi2 = j2[tid];
            float r1, r2; int ri1, ri2;
            const bool qb = (q1 > p1) || (q1 == p1 && qi1 < pi1);
            if (qb) {
                r1 = q1; ri1 = qi1;
                const bool pb = (p1 > q2) || (p1 == q2 && pi1 < qi2);
                if (pb) { r2 = p1; ri2 = pi1; } else { r2 = q2; ri2 = qi2; }
            } else {
                r1 = p1; ri1 = pi1;
                const bool qb2 = (q1 > p2) || (q1 == p2 && qi1 < pi2);
                if (qb2) { r2 = q1; ri2 = qi1; } else { r2 = p2; ri2 = pi2; }
            }
            s1[tid] = r1; j1[tid] = ri1; s2[tid] = r2; j2[tid] = ri2;
            mn[tid] = fminf(mn[tid], mn[tid + off]);
        }
        __syncthreads();
    }
    if (tid == 0) {
        n0[row] = j1[0];
        n1[row] = j2[0];
        const float t = 2.0f - 2.0f * mn[0];
        rowmax[row] = t * t;
    }
}

// ---------------------------------------------------------------------------
// k-reciprocal expansion for k1=1: per element a, compute idx4/vals rows.
// ---------------------------------------------------------------------------
__global__ void krecip(const float* __restrict__ S, const float* __restrict__ rowmax,
                       const int* __restrict__ n0, const int* __restrict__ n1,
                       int* __restrict__ idx4, float* __restrict__ vals, int A)
{
    const int a = blockIdx.x * blockDim.x + threadIdx.x;
    if (a >= A) return;
    const int c0 = n0[a], c1 = n1[a];
    const bool v0 = (n0[c0] == a) || (n1[c0] == a);
    const bool v1 = (n0[c1] == a) || (n1[c1] == a);
    const int e0 = n0[c0], e1 = n0[c1];
    const bool inR_e0 = (v0 && e0 == c0) || (v1 && e0 == c1);
    const bool inR_e1 = (v0 && e1 == c0) || (v1 && e1 == c1);
    const bool v2 = v0 && (n0[e0] == c0) && inR_e0;
    const bool v3 = v1 && (n0[e1] == c1) && inR_e1;

    int  id[4]  = {c0, c1, e0, e1};
    bool raw[4] = {v0, v1, v2, v3};
    bool kept[4]; kept[0] = v0;
    #pragma unroll
    for (int j = 1; j < 4; ++j) {
        bool dup = false;
        for (int k = 0; k < j; ++k) dup = dup || (kept[k] && id[k] == id[j]);
        kept[j] = raw[j] && !dup;
    }
    const float rm = rowmax[a];
    const float* Sr = S + (size_t)a * A;
    float w[4], ws = 0.f;
    #pragma unroll
    for (int k = 0; k < 4; ++k) {
        const float s = Sr[id[k]];
        const float t = 2.f - 2.f * s;
        const float d = t * t / rm;
        w[k] = kept[k] ? expf(-d) : 0.f;
        ws += w[k];
    }
    const float inv = 1.0f / (ws > 0.f ? ws : 1.0f);
    #pragma unroll
    for (int k = 0; k < 4; ++k) {
        vals[(size_t)a * 4 + k] = w[k] * inv;
        idx4[(size_t)a * 4 + k] = id[k];
    }
}

// ---------------------------------------------------------------------------
// Fused final stage: Jaccard (V never materialized: V[j,c]=sum_m vals[j,m]*[idx==c]),
// blended distance, and all three outputs.
// ---------------------------------------------------------------------------
__global__ __launch_bounds__(256)
void finalize(const float* __restrict__ S, const float* __restrict__ rowmax,
              const int* __restrict__ idx4, const float* __restrict__ vals,
              const float* __restrict__ iou, const float* __restrict__ kf,
              const float* __restrict__ thrp,
              float* __restrict__ out0, float* __restrict__ out1, float* __restrict__ out2)
{
    const size_t gid = (size_t)blockIdx.x * 256 + threadIdx.x;
    const int i  = (int)(gid / GG);
    const int j  = (int)(gid % GG);
    const int jj = QQ + j;

    int ii[4], ij[4]; float vi[4], vj[4];
    #pragma unroll
    for (int k = 0; k < 4; ++k) {
        ii[k] = idx4[(size_t)i * 4 + k];  vi[k] = vals[(size_t)i * 4 + k];
        ij[k] = idx4[(size_t)jj * 4 + k]; vj[k] = vals[(size_t)jj * 4 + k];
    }
    float t = 0.f;
    #pragma unroll
    for (int k = 0; k < 4; ++k) {
        float Vc = 0.f;
        #pragma unroll
        for (int m = 0; m < 4; ++m) Vc += (ij[m] == ii[k]) ? vj[m] : 0.f;
        t += fminf(vi[k], Vc);
    }
    const float jac = 1.0f - t / (2.0f - t);
    const float s   = S[(size_t)i * AA + jj];
    const float tt  = 2.0f - 2.0f * s;
    const float dq  = tt * tt / rowmax[i];
    const float fin = 0.7f * jac + 0.3f * dq;

    const size_t o = (size_t)i * GG + j;
    out0[o] = s;                 // Mp_before
    out1[o] = 1.0f - fin;        // Mp
    out2[o] = (kf[o] == -1.0f || iou[o] == 0.0f || s < thrp[0]) ? 1.0f : 0.0f;
}

// ---------------------------------------------------------------------------
extern "C" void kernel_launch(void* const* d_in, const int* in_sizes, int n_in,
                              void* d_out, int out_size, void* d_ws, size_t ws_size,
                              hipStream_t stream)
{
    (void)in_sizes; (void)n_in; (void)out_size; (void)ws_size;
    const float* Usrc = (const float*)d_in[0];   // (1, 512, 2048)
    const float* Utgt = (const float*)d_in[1];   // (1, 512, 2048)
    const float* iou  = (const float*)d_in[2];   // (2048, 2048)
    const float* kf   = (const float*)d_in[3];   // (2048, 2048)
    const float* thr  = (const float*)d_in[4];   // (1,)
    const float* cgw  = (const float*)d_in[5];   // (512, 512)
    const float* cgb  = (const float*)d_in[6];   // (512,)

    float* out  = (float*)d_out;
    float* out0 = out;                                   // Mp_before
    float* out1 = out + (size_t)QQ * GG;                 // Mp
    float* out2 = out + 2 * (size_t)QQ * GG;             // thr_flag

    // workspace layout
    float* S      = (float*)d_ws;                        // 4096*4096
    float* Mp0    = S      + (size_t)AA * AA;            // 2048*2048
    float* m1     = Mp0    + (size_t)N1 * N2;            // 2048*512
    float* m2     = m1     + (size_t)N1 * DD;            // 2048*512
    float* E      = m2     + (size_t)N2 * DD;            // 4096*512
    float* l1     = E      + (size_t)AA * DD;            // 2048
    float* l2     = l1     + N1;                         // 2048
    float* rowmax = l2     + N2;                         // 4096
    float* valsv  = rowmax + AA;                         // 4096*4
    int*   n0a    = (int*)(valsv + (size_t)AA * 4);      // 4096
    int*   n1a    = n0a + AA;                            // 4096
    int*   idx4a  = n1a + AA;                            // 4096*4

    const dim3 blk(256);

    // G1: Mp0 = Usrc^T * Utgt + iou          (M=N=2048, K=512)
    wmma_gemm<1,0,0,1,0><<<dim3(N2/64, N1/128), blk, 0, stream>>>(
        Usrc, N1, Utgt, N2, iou, N2, nullptr, Mp0, N2, DD);

    // G2: m1 = Mp0 * emb2 = Mp0 * Utgt^T     (M=2048, N=512, K=2048)
    wmma_gemm<0,1,0,0,0><<<dim3(DD/64, N1/128), blk, 0, stream>>>(
        Mp0, N2, Utgt, N2, nullptr, 0, nullptr, m1, DD, N2);

    // G3: m2 = Mp0^T * emb1 = Mp0^T * Usrc^T (M=2048, N=512, K=2048)
    wmma_gemm<1,1,0,0,0><<<dim3(DD/64, N2/128), blk, 0, stream>>>(
        Mp0, N2, Usrc, N1, nullptr, 0, nullptr, m2, DD, N1);

    // l1, l2 and x1 = emb1 + l1*m1 (in place over m1), likewise x2
    compute_l<<<dim3((N1 + 255) / 256), blk, 0, stream>>>(Usrc, m1, l1, N1, DD);
    compute_l<<<dim3((N2 + 255) / 256), blk, 0, stream>>>(Utgt, m2, l2, N2, DD);
    build_x<<<dim3((N1 * DD) / 256), blk, 0, stream>>>(Usrc, l1, m1, N1, DD);
    build_x<<<dim3((N2 * DD) / 256), blk, 0, stream>>>(Utgt, l2, m2, N2, DD);

    // G4/G5: E = relu(x * cg_w^T + cg_b)     (M=2048, N=512, K=512) each half
    wmma_gemm<0,1,1,0,1><<<dim3(DD/64, N1/128), blk, 0, stream>>>(
        m1, DD, cgw, DD, nullptr, 0, cgb, E, DD, DD);
    wmma_gemm<0,1,1,0,1><<<dim3(DD/64, N2/128), blk, 0, stream>>>(
        m2, DD, cgw, DD, nullptr, 0, cgb, E + (size_t)N1 * DD, DD, DD);

    // row-normalize E (eps 1e-12)
    row_norm<<<dim3(AA / 256), blk, 0, stream>>>(E, AA, DD);

    // G6: S = E * E^T (contains qq | qg / qg^T | gg blocks)  (M=N=4096, K=512)
    wmma_gemm<0,1,0,0,0><<<dim3(AA/64, AA/128), blk, 0, stream>>>(
        E, DD, E, DD, nullptr, 0, nullptr, S, AA, DD);

    // re-ranking: top-2 + rowmax, k-reciprocal rows, fused Jaccard + outputs
    row_top2<<<dim3(AA), blk, 0, stream>>>(S, AA, n0a, n1a, rowmax);
    krecip<<<dim3(AA / 256), blk, 0, stream>>>(S, rowmax, n0a, n1a, idx4a, valsv, AA);
    finalize<<<dim3(((size_t)QQ * GG) / 256), blk, 0, stream>>>(
        S, rowmax, idx4a, valsv, iou, kf, thr, out0, out1, out2);
}
